// ReCo_64098091926122
// MI455X (gfx1250) — matrix operand order repaired and checked by reading, not compiled, not run on hardware
//
#include <hip/hip_runtime.h>
#include <hip/hip_bf16.h>

// Problem constants (match reference)
#define HID    1024
#define GDIM   1024
#define BATCH  4096
#define MROWS  (BATCH * 4)   // 16384 rows of `pairs`/`cond`
#define KPAIR  2048          // K range fed by pairs (events)
#define KTOT   3072          // full K of cond
#define LDSTR  40            // LDS row stride in halfwords (32 data + 8 pad, keeps 16B align)
#define MTILE  128           // block tile M (wave tile 64x32, 2x2 waves)
#define NTILE  64            // block tile N

#if defined(__has_builtin) && __has_builtin(__builtin_amdgcn_sched_group_barrier)
#define SGB(mask, size, id) __builtin_amdgcn_sched_group_barrier(mask, size, id)
#else
#define SGB(mask, size, id)
#endif
// mask bits: 0x008 = MFMA/WMMA, 0x020 = VMEM read, 0x100 = DS read, 0x200 = DS write

typedef __attribute__((ext_vector_type(2)))  __bf16 v2bf;
typedef __attribute__((ext_vector_type(8)))  __bf16 v8bf;
typedef __attribute__((ext_vector_type(16))) __bf16 v16bf;
typedef __attribute__((ext_vector_type(8)))  float  v8f;
typedef __attribute__((ext_vector_type(2)))  unsigned u32x2;
typedef __attribute__((ext_vector_type(4)))  unsigned u32x4;

// Pack two f32 into packed bf16 (lowers to v_cvt_pk_bf16_f32)
__device__ __forceinline__ unsigned pk2(float a, float b) {
    v2bf v;
    v[0] = (__bf16)a;
    v[1] = (__bf16)b;
    return __builtin_bit_cast(unsigned, v);
}

// Assemble a 16-element bf16 fragment from two contiguous 16-byte LDS runs.
__device__ __forceinline__ v16bf ld_frag(const unsigned short* p, int hiOff) {
    v8bf lo = *(const v8bf*)(p);
    v8bf hi = *(const v8bf*)(p + hiOff);
    return __builtin_shufflevector(lo, hi, 0,1,2,3,4,5,6,7,8,9,10,11,12,13,14,15);
}

__global__ void init_ws_kernel(float* ws) {
    if (threadIdx.x < 8) ws[threadIdx.x] = 0.0f;
}

// Fused 4-way GEMM + bias + KL-term reduction.
// Block: 128 threads (4 waves, 2x2). Block tile 128(M) x 64(N). Wave tile 64x32.
// Register-pipelined global loads + double-buffered LDS (1 barrier / K-step) +
// sched_group_barrier-enforced [WMMA | VMEM | DSread] software pipeline.
__global__ __launch_bounds__(128)
void gemm_kl_kernel(const float* __restrict__ events,   // (B,5,H)
                    const float* __restrict__ contexts, // (B,4,H)
                    const float* __restrict__ Wz,  const float* __restrict__ Wv,
                    const float* __restrict__ Wqz, const float* __restrict__ Wqv,
                    const float* __restrict__ bz,  const float* __restrict__ bv,
                    const float* __restrict__ bqz, const float* __restrict__ bqv,
                    float* __restrict__ ws)
{
    __shared__ unsigned short ldsA[2][MTILE * LDSTR];       // 2 x 10240 B
    __shared__ unsigned short ldsB[2][4][NTILE * LDSTR];    // 2 x 20480 B

    const int tid   = threadIdx.x;
    const int lane  = tid & 31;
    const int wvid  = tid >> 5;      // 0..3
    const int waveM = wvid >> 1;     // 0..1  (owns 64 M-rows)
    const int waveN = wvid & 1;      // 0..1  (owns 32 N-cols)
    const int l15   = lane & 15;
    const int lhalf = lane >> 4;     // 0/1

    const int n0   = blockIdx.x * NTILE;   // N tile origin
    const int row0 = blockIdx.y * MTILE;   // M tile origin

    // A-staging geometry: 128 rows x 32 K, one thread per row (32 halfwords)
    const int sR = row0 + tid;
    const int sB = sR >> 2;
    const int sT = sR & 3;

    // B-staging geometry: each thread owns a 4(K) x 4(N) micro-block.
    const int n4  = (tid & 15) * 4;    // 0,4,...,60
    const int kk4 = (tid >> 4) * 4;    // 0,4,...,28

    const float* Ws4[4] = {Wz, Wv, Wqz, Wqv};

    v8f acc[4][2][4];                  // [mSub][nSub][{zm,zv,qm,qv}]
    const v8f vzero = {0.f,0.f,0.f,0.f,0.f,0.f,0.f,0.f};
    #pragma unroll
    for (int ms = 0; ms < 4; ++ms)
        #pragma unroll
        for (int ns = 0; ns < 2; ++ns)
            #pragma unroll
            for (int w = 0; w < 4; ++w) acc[ms][ns][w] = vzero;

    float4 aReg[8];        // prefetched A row slice (32 floats)
    float4 bReg[4][4];     // prefetched B micro-blocks, per weight

    // ---- helpers (inlined; `all4` is a literal at each call site) ----
    auto loadA_ev = [&](int kg) {      // kg multiple of 32; run never straddles the seam
        const int tt  = (kg >= HID) ? (sT + 1) : sT;
        const int col = kg - ((kg >= HID) ? HID : 0);
        const float* src = events + (((size_t)(sB * 5 + tt)) << 10) + col;
        #pragma unroll
        for (int i = 0; i < 8; ++i) aReg[i] = ((const float4*)src)[i];
    };
    auto loadA_cx = [&](int kg) {
        const float* src = contexts + (((size_t)(sB * 4 + sT)) << 10) + (kg - KPAIR);
        #pragma unroll
        for (int i = 0; i < 8; ++i) aReg[i] = ((const float4*)src)[i];
    };
    auto loadB = [&](int kg, bool all4) {
        #pragma unroll
        for (int w = 0; w < 4; ++w) {
            if (all4 || w >= 2) {
                const float* src = Ws4[w] + (size_t)(kg + kk4) * GDIM + n0 + n4;
                #pragma unroll
                for (int i = 0; i < 4; ++i)
                    bReg[w][i] = *(const float4*)(src + (size_t)i * GDIM);
            }
        }
    };
    auto storeA = [&](int p) {
        unsigned short* dst = &ldsA[p][tid * LDSTR];
        const float* f = (const float*)aReg;
        #pragma unroll
        for (int q = 0; q < 4; ++q) {
            u32x4 u = { pk2(f[q*8+0], f[q*8+1]), pk2(f[q*8+2], f[q*8+3]),
                        pk2(f[q*8+4], f[q*8+5]), pk2(f[q*8+6], f[q*8+7]) };
            ((u32x4*)dst)[q] = u;
        }
    };
    auto storeB = [&](int p, bool all4) {
        #pragma unroll
        for (int w = 0; w < 4; ++w) {
            if (all4 || w >= 2) {
                #pragma unroll
                for (int j = 0; j < 4; ++j) {
                    u32x2 u = { pk2(((const float*)&bReg[w][0])[j], ((const float*)&bReg[w][1])[j]),
                                pk2(((const float*)&bReg[w][2])[j], ((const float*)&bReg[w][3])[j]) };
                    *(u32x2*)&ldsB[p][w][(n4 + j) * LDSTR + kk4] = u;
                }
            }
        }
    };
    auto compute = [&](int p, bool all4) {
        v16bf afr[4];
        #pragma unroll
        for (int ms = 0; ms < 4; ++ms)
            afr[ms] = ld_frag(&ldsA[p][(waveM * 64 + ms * 16 + l15) * LDSTR + lhalf * 8], 16);

        auto ldB = [&](int ns, int w) -> v16bf {
            return ld_frag(&ldsB[p][w][(waveN * 32 + ns * 16 + l15) * LDSTR + lhalf * 8], 16);
        };
        auto doW = [&](int ns, int w, v16bf bfv) {
            #pragma unroll
            for (int ms = 0; ms < 4; ++ms)
                acc[ms][ns][w] = __builtin_amdgcn_wmma_f32_16x16x32_bf16(
                    false, afr[ms], false, bfv, (short)0, acc[ms][ns][w], false, false);
        };

        if (all4) {
            v16bf b0 = ldB(0, 0);
            v16bf b1 = ldB(0, 1); doW(0, 0, b0);
            v16bf b2 = ldB(0, 2); doW(0, 1, b1);
            v16bf b3 = ldB(0, 3); doW(0, 2, b2);
            v16bf b4 = ldB(1, 0); doW(0, 3, b3);
            v16bf b5 = ldB(1, 1); doW(1, 0, b4);
            v16bf b6 = ldB(1, 2); doW(1, 1, b5);
            v16bf b7 = ldB(1, 3); doW(1, 2, b6);
            doW(1, 3, b7);
        } else {
            v16bf b0 = ldB(0, 2);
            v16bf b1 = ldB(0, 3); doW(0, 2, b0);
            v16bf b2 = ldB(1, 2); doW(0, 3, b1);
            v16bf b3 = ldB(1, 3); doW(1, 2, b2);
            doW(1, 3, b3);
        }
    };
    // Pipeline shape for the branch-free steady-state body (phase 1):
    //   DSread x12 (A frags + b0,b1) ; 6x [WMMA x4 | VMEMread x4 | DSread x2] ;
    //   WMMA x8 ; DSwrite x20
    auto schedP1 = [&]() {
        SGB(0x100, 12, 0);
        #pragma unroll
        for (int g = 0; g < 6; ++g) {
            SGB(0x008, 4, 0);
            SGB(0x020, 4, 0);
            SGB(0x100, 2, 0);
        }
        SGB(0x008, 8, 0);
        SGB(0x200, 20, 0);
    };
    auto schedP2 = [&]() {
        SGB(0x100, 12, 0);
        #pragma unroll
        for (int g = 0; g < 2; ++g) {
            SGB(0x008, 4, 0);
            SGB(0x020, 8, 0);
            SGB(0x100, 2, 0);
        }
        SGB(0x008, 8, 0);
        SGB(0x200, 12, 0);
    };

    // ---------------- Phase 1: K in [0, 2048): A from events, all 4 weights
    loadA_ev(0);
    loadB(0, true);
    storeA(0);
    storeB(0, true);
    __syncthreads();
    for (int ks = 0; ks < KPAIR / 32 - 1; ++ks) {   // branch-free steady state
        const int p = ks & 1;
        loadA_ev((ks + 1) * 32);
        loadB((ks + 1) * 32, true);
        compute(p, true);
        storeA(p ^ 1);
        storeB(p ^ 1, true);
        schedP1();
        __syncthreads();
    }
    // Last phase-1 tile overlapped with phase-2 prologue staging (opposite buffer).
    loadA_cx(KPAIR);
    loadB(KPAIR, false);
    compute((KPAIR / 32 - 1) & 1, true);            // buffer 1
    storeA(0);
    storeB(0, false);
    __syncthreads();

    // ---------------- Phase 2: K in [2048, 3072): A from contexts, q-weights only
    for (int ks2 = 0; ks2 < (KTOT - KPAIR) / 32 - 1; ++ks2) {
        const int p = ks2 & 1;
        loadA_cx(KPAIR + (ks2 + 1) * 32);
        loadB(KPAIR + (ks2 + 1) * 32, false);
        compute(p, false);
        storeA(p ^ 1);
        storeB(p ^ 1, false);
        schedP2();
        __syncthreads();
    }
    compute(((KTOT - KPAIR) / 32 - 1) & 1, false);  // final tile, no prefetch

    // ---------------- Epilogue: bias + KL elementwise term, reduced in registers.
    // C/D layout: VGPR j holds M = j + 8*(lane>=16), N = lane%16 -> bias depends only on lane%16.
    float local = 0.0f;
    #pragma unroll
    for (int ms = 0; ms < 4; ++ms) {
        #pragma unroll
        for (int ns = 0; ns < 2; ++ns) {
            const int n = n0 + waveN * 32 + ns * 16 + l15;
            const float vbz  = bz[n],  vbv  = bv[n];
            const float vbqz = bqz[n], vbqv = bqv[n];
            #pragma unroll
            for (int j = 0; j < 8; ++j) {
                float zm = acc[ms][ns][0][j] + vbz;
                float zv = acc[ms][ns][1][j] + vbv;
                float qm = acc[ms][ns][2][j] + vbqz;
                float qv = acc[ms][ns][3][j] + vbqv;
                float d  = zm - qm;
                local += qv - zv + (__expf(zv) + d * d) * __expf(-qv) - 1.0f;
            }
        }
    }
    #pragma unroll
    for (int off = 16; off > 0; off >>= 1) local += __shfl_xor(local, off);
    if (lane == 0) atomicAdd(&ws[0], local);
}

// Tail: cross-entropy + logic terms over (B,3,2) tensors. One thread per batch row.
__global__ __launch_bounds__(256)
void small_loss_kernel(const float* __restrict__ ps, const float* __restrict__ pt,
                       const float* __restrict__ pc,
                       const int* __restrict__ labels, const int* __restrict__ scene,
                       const int* __restrict__ thresh, float* __restrict__ ws)
{
    const int b = blockIdx.x * blockDim.x + threadIdx.x;
    float ce_num = 0.f, ce_den = 0.f, lg_num = 0.f, lg_den = 0.f;
    if (b < BATCH) {
        const int lab = labels[b], s = scene[b], t = thresh[b];
        const bool l5 = (lab == 5), l2 = (lab == 2), l3 = (lab == 3);
        const bool lelse = !(l5 || l2 || l3);
        const int n_valid = l5 ? 3 : (l2 ? 1 : (l3 ? 2 : 3));
        const int last = n_valid - 1;
        #pragma unroll
        for (int slot = 0; slot < 3; ++slot) {
            const bool valid = slot < n_valid;
            const int chain = l5 ? 1 : (l2 ? 0 : (l3 ? (slot == 0 ? 1 : 0)
                                                     : (slot < 2 ? 1 : 0)));
            const int ps_col  = (!l5 && slot == last) ? s : 0;
            const int pt_row  = (l3 && slot == 1) ? ((t == 1) ? 0 : 1) : slot;
            const int pt_col  = (l2 && slot == 0) ? t : ((lelse && slot == 2) ? t : 0);
            const int pcs_col = l5 ? 1 : ((slot == last) ? 0 : 1);

            const float PS  = ps[(b * 3 + slot)   * 2 + ps_col];
            const float PT  = pt[(b * 3 + pt_row) * 2 + pt_col];
            const float PCS = pc[(b * 3 + slot)   * 2 + pcs_col];
            const float p0  = pc[(b * 3 + slot) * 2 + 0];
            const float p1  = pc[(b * 3 + slot) * 2 + 1];

            const float m   = fmaxf(p0, p1);
            const float lse = m + __logf(__expf(p0 - m) + __expf(p1 - m));
            const float ce  = -(((chain == 0) ? p0 : p1) - lse);
            const float w   = ((chain == 0) ? 0.6f : 0.4f) * (valid ? 1.f : 0.f);
            ce_num += w * ce;
            ce_den += w;

            const float lsig = fminf(PCS, 0.f) - log1pf(__expf(-fabsf(PCS)));
            const float term = fabsf(__logf(PS) + __logf(PT) - lsig);
            if (valid) { lg_num += term; lg_den += 1.f; }
        }
    }
    #pragma unroll
    for (int off = 16; off > 0; off >>= 1) {
        ce_num += __shfl_xor(ce_num, off);
        ce_den += __shfl_xor(ce_den, off);
        lg_num += __shfl_xor(lg_num, off);
        lg_den += __shfl_xor(lg_den, off);
    }
    if ((threadIdx.x & 31) == 0) {
        atomicAdd(&ws[1], ce_num);
        atomicAdd(&ws[2], ce_den);
        atomicAdd(&ws[3], lg_num);
        atomicAdd(&ws[4], lg_den);
    }
}

__global__ void combine_kernel(const float* __restrict__ ws, float* __restrict__ out) {
    if (threadIdx.x == 0 && blockIdx.x == 0) {
        const float kl_loss = 0.5f * ws[0] / (float)MROWS;   // kl = 0.5*sum ; /(B*4)
        out[0] = ws[1] / ws[2] + kl_loss + ws[3] / ws[4];    // ce + L2*kl + L1*logic
    }
}

extern "C" void kernel_launch(void* const* d_in, const int* in_sizes, int n_in,
                              void* d_out, int out_size, void* d_ws, size_t ws_size,
                              hipStream_t stream) {
    const float* events   = (const float*)d_in[0];
    const float* contexts = (const float*)d_in[1];
    const float* ps       = (const float*)d_in[2];
    const float* pt       = (const float*)d_in[3];
    const float* pc       = (const float*)d_in[4];
    const int*   labels   = (const int*)d_in[5];
    const int*   scene    = (const int*)d_in[6];
    const int*   thresh   = (const int*)d_in[7];
    const float* Wz       = (const float*)d_in[8];
    const float* bz       = (const float*)d_in[9];
    const float* Wv       = (const float*)d_in[10];
    const float* bv       = (const float*)d_in[11];
    const float* Wqz      = (const float*)d_in[12];
    const float* bqz      = (const float*)d_in[13];
    const float* Wqv      = (const float*)d_in[14];
    const float* bqv      = (const float*)d_in[15];

    float* ws  = (float*)d_ws;
    float* out = (float*)d_out;

    init_ws_kernel<<<1, 32, 0, stream>>>(ws);

    dim3 grid(GDIM / NTILE, MROWS / MTILE);   // 16 x 128 blocks
    gemm_kl_kernel<<<grid, 128, 0, stream>>>(events, contexts,
                                             Wz, Wv, Wqz, Wqv,
                                             bz, bv, bqz, bqv, ws);

    small_loss_kernel<<<BATCH / 256, 256, 0, stream>>>(ps, pt, pc, labels, scene, thresh, ws);

    combine_kernel<<<1, 1, 0, stream>>>(ws, out);
}
